// MSAPairWeightedAveraging_15668040696174
// MI455X (gfx1250) — matrix operand order repaired and checked by reading, not compiled, not run on hardware
//
#include <hip/hip_runtime.h>
#include <hip/hip_bf16.h>

typedef __bf16 bf16;
typedef __attribute__((ext_vector_type(16))) __bf16 v16bf;
typedef __attribute__((ext_vector_type(8)))  __bf16 v8bf;
typedef __attribute__((ext_vector_type(8)))  float  v8f;

#define NSEQ 256
#define NRES 512
#define CMSA 64
#define CZ   128
#define CHID 32
#define NH   8
#define HC   256   // NH*CHID
#define INFV 1e8f
#define EPSV 1e-5f

// ---------------------------------------------------------------------------
// WMMA fragment tiles are stored PRE-SWIZZLED: a 16x16-output tile's operand
// tile occupies 512 bf16 (1 KB); lane l's 16 fragment elements live at
// tile[l*16 .. l*16+15] (contiguous 32 B -> 2x b128 loads, no gathers).
// Slot maps use contiguous k order (k = 16*(lane/16) + e) for A and B alike:
//   A (16 M x 32 K): element (m,k) -> lane = m + (k&16 ? 16:0), e = k & 15
//   B (32 K x 16 N): element (k,n) -> lane = (k&16 ? 16:0) + n, e = k & 15
// ---------------------------------------------------------------------------

__device__ __forceinline__ int a_slot(int m, int k32) {
  return (m + ((k32 & 16) ? 16 : 0)) * 16 + (k32 & 15);
}
__device__ __forceinline__ int b_slot(int k32, int n16) {
  return (((k32 & 16) ? 16 : 0) + n16) * 16 + (k32 & 15);
}

__device__ __forceinline__ v8f wmma_bf16(v16bf a, v16bf b, v8f c) {
  return __builtin_amdgcn_wmma_f32_16x16x32_bf16(
      /*neg_a=*/false, a, /*neg_b=*/false, b,
      /*c_mod=*/(short)0, c, /*reuse_a=*/false, /*reuse_b=*/false);
}

__device__ __forceinline__ v16bf frag_ld(const bf16* p) {
  return *(const v16bf*)p;   // 32 B contiguous per lane
}

__device__ __forceinline__ float fast_sigmoid(float x) {
  // v_exp_f32 + v_add + v_rcp_f32 (no IEEE division expansion)
  return __builtin_amdgcn_rcpf(1.f + __expf(-x));
}

// C/D fragment: element e -> M = e + (lane>=16 ? 8 : 0), N = lane & 15

// ---------- Kernel 0: weights -> bf16, pre-swizzled B-fragment tiles ----------
// wv_sw/wg_sw: [kc(2)][ntile(16)] tiles (K=64, N=256)
// wot_sw:      [kc(8)][ntile(4)]  tiles (K=256, N=64)
__global__ __launch_bounds__(256) void k_prep(const float* __restrict__ Wv,
                                              const float* __restrict__ Wg,
                                              const float* __restrict__ Wo,
                                              bf16* __restrict__ wv_sw,
                                              bf16* __restrict__ wg_sw,
                                              bf16* __restrict__ wot_sw) {
  int idx = blockIdx.x * 256 + threadIdx.x;   // 0..16383
  {
    int n = idx >> 6, k = idx & 63;           // W_v[n][k], W_g[n][k]
    int off = ((k >> 5) * 16 + (n >> 4)) * 512 + b_slot(k & 31, n & 15);
    wv_sw[off] = (bf16)Wv[idx];
    wg_sw[off] = (bf16)Wg[idx];
  }
  {
    int n = idx >> 8, k = idx & 255;          // W_o[n][k]
    int off = ((k >> 5) * 4 + (n >> 4)) * 512 + b_slot(k & 31, n & 15);
    wot_sw[off] = (bf16)Wo[idx];
  }
}

// ---------- Kernel 1: LayerNorm(m) -> v (masked, B-tiles) and g (sigmoid, A-tiles) ----------
// grid: 8192 blocks of 16 rows; 128 threads = 4 waves.
// v_sw layout: [s][h][jchunk(16)][nt(2)] B-tiles for the einsum GEMM.
// g_sw layout: [rowtile(8192)][kchunk(8)] A-tiles matching o_sw (final GEMM A).
__global__ __launch_bounds__(128) void k_ln_vg(const float* __restrict__ m,
                                               const float* __restrict__ msa_mask,
                                               const float* __restrict__ lng,
                                               const float* __restrict__ lnb,
                                               const bf16* __restrict__ wv_sw,
                                               const bf16* __restrict__ wg_sw,
                                               bf16* __restrict__ v_sw,
                                               bf16* __restrict__ g_sw) {
  __shared__ float mrows[16 * 64];
  __shared__ float mu_s[16], rs_s[16];
  __shared__ __align__(32) bf16 asw[2][512];   // A-fragment tiles, K chunks 0/1

  const int rowbase = blockIdx.x * 16;
  const int tid = threadIdx.x, wave = tid >> 5, lane = tid & 31;

  for (int idx = tid; idx < 1024; idx += 128)
    mrows[idx] = m[(size_t)rowbase * 64 + idx];
  __syncthreads();

  if (tid < 16) {
    float sum = 0.f, sq = 0.f;
    for (int c = 0; c < 64; ++c) { float x = mrows[tid * 64 + c]; sum += x; sq += x * x; }
    float mu  = sum * (1.f / 64.f);
    float var = sq * (1.f / 64.f) - mu * mu;
    mu_s[tid] = mu;
    rs_s[tid] = rsqrtf(var + EPSV);
  }
  __syncthreads();

  for (int idx = tid; idx < 1024; idx += 128) {
    int r = idx >> 6, c = idx & 63;
    float val = (mrows[idx] - mu_s[r]) * rs_s[r] * lng[c] + lnb[c];
    asw[c >> 5][a_slot(r, c & 31)] = (bf16)val;
  }
  __syncthreads();

  const v16bf a0 = frag_ld(&asw[0][lane * 16]);
  const v16bf a1 = frag_ld(&asw[1][lane * 16]);

  const int s      = rowbase >> 9;
  const int j      = rowbase & (NRES - 1);
  const int jchunk = j >> 5;
  const int jhalf  = (j >> 4) & 1;

  if (wave < 2) {
    // ---- v path: 8 tiles per wave, masked, packed b128 stores into B-tiles ----
    float mk[8];
#pragma unroll
    for (int e = 0; e < 8; ++e)
      mk[e] = msa_mask[(size_t)rowbase + e + ((lane >> 4) << 3)];

    for (int tt = 0; tt < 8; ++tt) {
      int n0 = (wave * 8 + tt) * 16;          // 0..240 step 16 over both waves
      int h  = n0 >> 5, nt = (n0 >> 4) & 1;
      v8f acc = {};
      acc = wmma_bf16(a0, frag_ld(&wv_sw[(0 * 16 + (n0 >> 4)) * 512 + lane * 16]), acc);
      acc = wmma_bf16(a1, frag_ld(&wv_sw[(1 * 16 + (n0 >> 4)) * 512 + lane * 16]), acc);

      v8bf pv;
#pragma unroll
      for (int e = 0; e < 8; ++e) pv[e] = (bf16)(acc[e] * mk[e]);

      size_t toff = ((((size_t)s * NH + h) * 16 + jchunk) * 2 + nt) * 512;
      int slot = ((jhalf ? 16 : 0) + (lane & 15)) * 16 + ((lane >> 4) << 3);
      *(v8bf*)&v_sw[toff + slot] = pv;
    }
  } else {
    // ---- g path: sigmoid, scatter into A-tiles (final-GEMM layout) ----
    size_t rowtile = (size_t)rowbase >> 4;
    for (int tt = 0; tt < 8; ++tt) {
      int n0 = ((wave - 2) * 8 + tt) * 16;
      v8f acc = {};
      acc = wmma_bf16(a0, frag_ld(&wg_sw[(0 * 16 + (n0 >> 4)) * 512 + lane * 16]), acc);
      acc = wmma_bf16(a1, frag_ld(&wg_sw[(1 * 16 + (n0 >> 4)) * 512 + lane * 16]), acc);

      int    kchunk = n0 >> 5;                 // head index
      int    k32    = (n0 & 16) + (lane & 15);
      size_t base   = (rowtile * 8 + kchunk) * 512;
#pragma unroll
      for (int e = 0; e < 8; ++e) {
        int mm = e + ((lane >> 4) << 3);
        g_sw[base + a_slot(mm, k32)] = (bf16)fast_sigmoid(acc[e]);
      }
    }
  }
}

// ---------- Kernel 2: b = LN(z)@W_b^T + mask; softmax over j -> A-tiles ----------
// wA layout: [h][itile(32)][kc(16)] A-tiles (M=i, K=j) for the einsum GEMM.
__global__ __launch_bounds__(256) void k_bias_softmax(const float* __restrict__ z,
                                                      const float* __restrict__ z_mask,
                                                      const float* __restrict__ lng,
                                                      const float* __restrict__ lnb,
                                                      const float* __restrict__ Wb,
                                                      bf16* __restrict__ wA) {
  __shared__ float bsm[NRES * NH];   // 16 KB
  const int i = blockIdx.x;
  const int tid = threadIdx.x, wave = tid >> 5, lane = tid & 31;

  for (int j = wave; j < NRES; j += 8) {
    const float* zr = z + ((size_t)i * NRES + j) * CZ;
    float x[4];
    float sum = 0.f, sq = 0.f;
#pragma unroll
    for (int q = 0; q < 4; ++q) {
      x[q] = zr[lane * 4 + q];
      sum += x[q]; sq += x[q] * x[q];
    }
#pragma unroll
    for (int o = 16; o > 0; o >>= 1) {
      sum += __shfl_xor(sum, o, 32);
      sq  += __shfl_xor(sq,  o, 32);
    }
    float mu  = sum * (1.f / 128.f);
    float var = sq * (1.f / 128.f) - mu * mu;
    float rs  = rsqrtf(var + EPSV);

    float acc[NH];
#pragma unroll
    for (int h = 0; h < NH; ++h) acc[h] = 0.f;
#pragma unroll
    for (int q = 0; q < 4; ++q) {
      int   c  = lane * 4 + q;
      float nx = (x[q] - mu) * rs * lng[c] + lnb[c];
#pragma unroll
      for (int h = 0; h < NH; ++h) acc[h] += nx * Wb[h * CZ + c];
    }
#pragma unroll
    for (int h = 0; h < NH; ++h)
#pragma unroll
      for (int o = 16; o > 0; o >>= 1) acc[h] += __shfl_xor(acc[h], o, 32);

    if (lane == 0) {
      float mterm = INFV * (z_mask[(size_t)i * NRES + j] - 1.f);
#pragma unroll
      for (int h = 0; h < NH; ++h) bsm[j * NH + h] = acc[h] + mterm;
    }
  }
  __syncthreads();

  // softmax over j, one wave per head; store into pre-swizzled A-tiles
  const int h = wave;
  float mx = -3.4e38f;
  for (int j = lane; j < NRES; j += 32) mx = fmaxf(mx, bsm[j * NH + h]);
#pragma unroll
  for (int o = 16; o > 0; o >>= 1) mx = fmaxf(mx, __shfl_xor(mx, o, 32));
  float s = 0.f;
  for (int j = lane; j < NRES; j += 32) s += __expf(bsm[j * NH + h] - mx);
#pragma unroll
  for (int o = 16; o > 0; o >>= 1) s += __shfl_xor(s, o, 32);
  float inv = __builtin_amdgcn_rcpf(s);

  const int mrow = i & 15, itile = i >> 4;
  bf16* wbase = wA + (size_t)(h * 32 + itile) * 16 * 512;
  for (int j = lane; j < NRES; j += 32) {
    float wv = __expf(bsm[j * NH + h] - mx) * inv;
    wbase[(size_t)(j >> 5) * 512 + a_slot(mrow, j & 31)] = (bf16)wv;
  }
}

// ---------- Kernel 3: per-head einsum + gate; no LDS, 2x2 register blocking ----------
// grid: (s=256, it=2, h=8); 256 threads = 8 waves; wave -> 2 i-tiles x 2 N-tiles
// Per K-step: 2 A-frags + 2 B-frags (8x b128) feed 4 independent WMMAs.
__global__ __launch_bounds__(256) void k_attn(const bf16* __restrict__ wA,
                                              const bf16* __restrict__ v_sw,
                                              const bf16* __restrict__ g_sw,
                                              bf16* __restrict__ o_sw) {
  const int s = blockIdx.x, it = blockIdx.y, h = blockIdx.z;
  const int tid = threadIdx.x, wave = tid >> 5, lane = tid & 31;
  const int itile0 = it * 16 + wave * 2;           // 0..30 step 2
  const int itile1 = itile0 + 1;

  const bf16* abase0 = wA   + ((size_t)(h * 32 + itile0) * 16) * 512 + lane * 16;
  const bf16* abase1 = wA   + ((size_t)(h * 32 + itile1) * 16) * 512 + lane * 16;
  const bf16* vbase  = v_sw + (((size_t)s * NH + h) * 16) * 2 * 512 + lane * 16;

  v8f acc00 = {}, acc01 = {}, acc10 = {}, acc11 = {};
#pragma unroll 4
  for (int kc = 0; kc < 16; ++kc) {
    v16bf a0 = frag_ld(abase0 + (size_t)kc * 512);
    v16bf a1 = frag_ld(abase1 + (size_t)kc * 512);
    v16bf b0 = frag_ld(vbase + ((size_t)kc * 2 + 0) * 512);
    v16bf b1 = frag_ld(vbase + ((size_t)kc * 2 + 1) * 512);
    acc00 = wmma_bf16(a0, b0, acc00);
    acc01 = wmma_bf16(a0, b1, acc01);
    acc10 = wmma_bf16(a1, b0, acc10);
    acc11 = wmma_bf16(a1, b1, acc11);
  }

  // epilogue: gate-multiply and store into A-tiles for the final GEMM
  const int k32a = lane & 15;        // acc*0 -> K 0..15 of the head chunk
  const int k32b = 16 + (lane & 15); // acc*1 -> K 16..31
#pragma unroll
  for (int t = 0; t < 2; ++t) {
    const v8f ca = t ? acc10 : acc00;
    const v8f cb = t ? acc11 : acc01;
    const int itile = t ? itile1 : itile0;
    const size_t rowtile = ((size_t)s * NRES + itile * 16) >> 4;
    const size_t base    = (rowtile * 8 + h) * 512;
#pragma unroll
    for (int e = 0; e < 8; ++e) {
      int mm = e + ((lane >> 4) << 3);
      int sa = a_slot(mm, k32a);
      int sb = a_slot(mm, k32b);
      float g0 = (float)g_sw[base + sa];
      float g1 = (float)g_sw[base + sb];
      o_sw[base + sa] = (bf16)(ca[e] * g0);
      o_sw[base + sb] = (bf16)(cb[e] * g1);
    }
  }
}

// ---------- Kernel 4: out = o @ W_o^T (f32), no LDS ----------
// grid: 8192 row-tiles; 128 threads = 4 waves, one 16-col N-tile each
__global__ __launch_bounds__(128) void k_outproj(const bf16* __restrict__ o_sw,
                                                 const bf16* __restrict__ wot_sw,
                                                 float* __restrict__ out) {
  const int rowtile = blockIdx.x;
  const int tid = threadIdx.x, wave = tid >> 5, lane = tid & 31;

  const bf16* abase = o_sw + (size_t)rowtile * 8 * 512 + lane * 16;
  v8f acc = {};
#pragma unroll
  for (int kc = 0; kc < 8; ++kc) {
    v16bf a = frag_ld(abase + (size_t)kc * 512);
    v16bf b = frag_ld(&wot_sw[(size_t)(kc * 4 + wave) * 512 + lane * 16]);
    acc = wmma_bf16(a, b, acc);
  }
  const int n = wave * 16 + (lane & 15);
#pragma unroll
  for (int e = 0; e < 8; ++e) {
    int mr = e + ((lane >> 4) << 3);
    out[(size_t)(rowtile * 16 + mr) * CMSA + n] = acc[e];
  }
}

// ---------- launch ----------
extern "C" void kernel_launch(void* const* d_in, const int* in_sizes, int n_in,
                              void* d_out, int out_size, void* d_ws, size_t ws_size,
                              hipStream_t stream) {
  const float* m        = (const float*)d_in[0];
  const float* z        = (const float*)d_in[1];
  const float* msa_mask = (const float*)d_in[2];
  const float* z_mask   = (const float*)d_in[3];
  const float* ln_m_g   = (const float*)d_in[4];
  const float* ln_m_b   = (const float*)d_in[5];
  const float* W_v      = (const float*)d_in[6];
  const float* W_g      = (const float*)d_in[7];
  const float* ln_z_g   = (const float*)d_in[8];
  const float* ln_z_b   = (const float*)d_in[9];
  const float* W_b      = (const float*)d_in[10];
  const float* W_o      = (const float*)d_in[11];
  float* out            = (float*)d_out;

  char* ws = (char*)d_ws;
  const size_t SZ_W16K = (size_t)16384 * sizeof(bf16);              // 32 KB each
  const size_t SZ_V    = (size_t)NSEQ * NRES * HC * sizeof(bf16);   // 64 MB
  const size_t SZ_WA   = (size_t)NH * NRES * NRES * sizeof(bf16);   // 4 MB

  bf16* wv_sw  = (bf16*)(ws);
  bf16* wg_sw  = (bf16*)(ws + SZ_W16K);
  bf16* wot_sw = (bf16*)(ws + 2 * SZ_W16K);
  bf16* v_sw   = (bf16*)(ws + 3 * SZ_W16K);
  bf16* g_sw   = (bf16*)((char*)v_sw + SZ_V);
  bf16* wA     = (bf16*)((char*)g_sw + SZ_V);
  bf16* o_sw   = (bf16*)((char*)wA + SZ_WA);

  k_prep<<<64, 256, 0, stream>>>(W_v, W_g, W_o, wv_sw, wg_sw, wot_sw);

  k_ln_vg<<<(NSEQ * NRES) / 16, 128, 0, stream>>>(m, msa_mask, ln_m_g, ln_m_b,
                                                  wv_sw, wg_sw, v_sw, g_sw);

  k_bias_softmax<<<NRES, 256, 0, stream>>>(z, z_mask, ln_z_g, ln_z_b, W_b, wA);

  dim3 g3(NSEQ, 2, NH);
  k_attn<<<g3, 256, 0, stream>>>(wA, v_sw, g_sw, o_sw);

  k_outproj<<<(NSEQ * NRES) / 16, 128, 0, stream>>>(o_sw, wot_sw, out);
}
